// FP8LinearInference_54193897341244
// MI455X (gfx1250) — 8-bit matrix operand order repaired and checked by reading, compiled, not run on hardware
//
#include <hip/hip_runtime.h>
#include <hip/hip_bf16.h>
#include <stdint.h>

typedef int   v16i __attribute__((ext_vector_type(16)));
typedef float v8f  __attribute__((ext_vector_type(8)));

#define F8_MAX 448.0f
#define QEPS   1e-12f

__global__ void __launch_bounds__(256)
fp8lin_amax_kernel(const float* __restrict__ src, size_t n4,
                   unsigned* __restrict__ dst_bits) {
  const float4* p = reinterpret_cast<const float4*>(src);
  size_t i      = (size_t)blockIdx.x * blockDim.x + threadIdx.x;
  size_t stride = (size_t)gridDim.x * blockDim.x;
  float m = 0.0f;
  for (; i < n4; i += stride) {
    float4 v = p[i];
    m = fmaxf(m, fmaxf(fmaxf(fabsf(v.x), fabsf(v.y)),
                       fmaxf(fabsf(v.z), fabsf(v.w))));
  }
  #pragma unroll
  for (int off = 16; off > 0; off >>= 1)
    m = fmaxf(m, __shfl_xor(m, off, 32));
  if ((threadIdx.x & 31) == 0)
    atomicMax(dst_bits, __float_as_uint(m));
}

__global__ void __launch_bounds__(256)
fp8lin_quant_kernel(const float* __restrict__ src, size_t n8,
                    const unsigned* __restrict__ amax_bits,
                    uint8_t* __restrict__ dst) {
  size_t i = (size_t)blockIdx.x * blockDim.x + threadIdx.x;
  if (i >= n8) return;
  float ax = __uint_as_float(*amax_bits);
  float rs = F8_MAX / fmaxf(ax, QEPS);
  const float4* p = reinterpret_cast<const float4*>(src) + 2 * i;
  float4 v0 = p[0];
  float4 v1 = p[1];
  int lo = 0, hi = 0;
  lo = __builtin_amdgcn_cvt_pk_fp8_f32(v0.x * rs, v0.y * rs, lo, false);
  lo = __builtin_amdgcn_cvt_pk_fp8_f32(v0.z * rs, v0.w * rs, lo, true);
  hi = __builtin_amdgcn_cvt_pk_fp8_f32(v1.x * rs, v1.y * rs, hi, false);
  hi = __builtin_amdgcn_cvt_pk_fp8_f32(v1.z * rs, v1.w * rs, hi, true);
  reinterpret_cast<int2*>(dst)[i] = make_int2(lo, hi);
}


__device__ __forceinline__ void fp8lin_load_a(v16i& a, const int2* base, int kb) {
  const int2* p = base + (kb >> 3);
  #pragma unroll
  for (int j = 0; j < 8; ++j) {
    int2 t = p[4 * (j >> 1) + (j & 1)];
    a[2 * j]     = t.x;
    a[2 * j + 1] = t.y;
  }
}

__device__ __forceinline__ void fp8lin_load_b(v16i& b, const uint8_t* rowbase, int kb) {
  const int4* p = reinterpret_cast<const int4*>(rowbase + kb);
  #pragma unroll
  for (int t = 0; t < 4; ++t) {
    int4 q = p[2 * t];
    b[4 * t]     = q.x;
    b[4 * t + 1] = q.y;
    b[4 * t + 2] = q.z;
    b[4 * t + 3] = q.w;
  }
}

__device__ __forceinline__ void fp8lin_mma4(v8f acc[2][2],
                                            const v16i a[2], const v16i b[2]) {
  #pragma unroll
  for (int i = 0; i < 2; ++i)
    #pragma unroll
    for (int j = 0; j < 2; ++j)
      acc[i][j] = __builtin_amdgcn_wmma_f32_16x16x128_fp8_fp8(
          a[i], b[j], (short)0, acc[i][j], false, false);
}

__global__ void __launch_bounds__(256)
fp8lin_gemm_kernel(const uint8_t* __restrict__ x8,
                   const uint8_t* __restrict__ w8,
                   const float* __restrict__ bias,
                   const unsigned* __restrict__ hdr,
                   float* __restrict__ out,
                   int M, int N, int K) {
  const int lane = threadIdx.x & 31;
  const int wave = threadIdx.x >> 5;

  const int nTiles = N >> 5;
  const int tile   = blockIdx.x * 8 + wave;
  const int mT = tile / nTiles;
  const int nT = tile % nTiles;
  const int m0 = mT << 5;
  const int n0 = nT << 5;
  if (m0 >= M) return;

  const float ax = __uint_as_float(hdr[0]);
  const float aw = __uint_as_float(hdr[1]);
  const float s  = (fmaxf(ax, QEPS) / F8_MAX) * (fmaxf(aw, QEPS) / F8_MAX);

  const int row  = lane & 15;
  const int hsel = lane >> 4;

  const int2* pa0 = reinterpret_cast<const int2*>(
      x8 + (size_t)(m0 + row) * K + (hsel ? 16 : 0));
  const int2* pa1 = reinterpret_cast<const int2*>(
      x8 + (size_t)(m0 + 16 + row) * K + (hsel ? 16 : 0));
  const uint8_t* pb0 = w8 + (size_t)(n0 + row) * K + (hsel ? 16 : 0);
  const uint8_t* pb1 = w8 + (size_t)(n0 + 16 + row) * K + (hsel ? 16 : 0);

  v8f acc[2][2] = {};

  v16i a0[2], b0[2];
  v16i a1[2], b1[2];

  fp8lin_load_a(a0[0], pa0, 0);
  fp8lin_load_a(a0[1], pa1, 0);
  fp8lin_load_b(b0[0], pb0, 0);
  fp8lin_load_b(b0[1], pb1, 0);

  for (int kb = 0; kb < K; kb += 256) {
    fp8lin_load_a(a1[0], pa0, kb + 128);
    fp8lin_load_a(a1[1], pa1, kb + 128);
    fp8lin_load_b(b1[0], pb0, kb + 128);
    fp8lin_load_b(b1[1], pb1, kb + 128);

    fp8lin_mma4(acc, a0, b0);

    const int kn = kb + 256;
    if (kn < K) {
      fp8lin_load_a(a0[0], pa0, kn);
      fp8lin_load_a(a0[1], pa1, kn);
      fp8lin_load_b(b0[0], pb0, kn);
      fp8lin_load_b(b0[1], pb1, kn);
    }

    fp8lin_mma4(acc, a1, b1);
  }

  #pragma unroll
  for (int j = 0; j < 2; ++j) {
    const int col  = n0 + 16 * j + row;
    const float bv = bias[col];
    #pragma unroll
    for (int i = 0; i < 2; ++i) {
      const int mrow = m0 + 16 * i + (hsel ? 8 : 0);
      #pragma unroll
      for (int r = 0; r < 8; ++r) {
        out[(size_t)(mrow + r) * N + col] = acc[i][j][r] * s + bv;
      }
    }
  }
}

extern "C" void kernel_launch(void* const* d_in, const int* in_sizes, int n_in,
                              void* d_out, int out_size, void* d_ws, size_t ws_size,
                              hipStream_t stream) {
  const float* x    = (const float*)d_in[0];
  const float* w    = (const float*)d_in[1];
  const float* bias = (const float*)d_in[2];
  float* out = (float*)d_out;

  const int N = in_sizes[2];
  const int K = in_sizes[1] / N;
  const int M = in_sizes[0] / K;

  uint8_t*  ws  = (uint8_t*)d_ws;
  unsigned* hdr = (unsigned*)ws;
  uint8_t*  x8  = ws + 256;
  uint8_t*  w8  = x8 + (size_t)M * K;

  hipMemsetAsync(d_ws, 0, 256, stream);

  const size_t nx = (size_t)M * K;
  const size_t nw = (size_t)N * K;

  fp8lin_amax_kernel<<<2048, 256, 0, stream>>>(x, nx / 4, hdr + 0);
  fp8lin_amax_kernel<<<256,  256, 0, stream>>>(w, nw / 4, hdr + 1);

  fp8lin_quant_kernel<<<(unsigned)((nx / 8 + 255) / 256), 256, 0, stream>>>(
      x, nx / 8, hdr + 0, x8);
  fp8lin_quant_kernel<<<(unsigned)((nw / 8 + 255) / 256), 256, 0, stream>>>(
      w, nw / 8, hdr + 1, w8);

  const int tiles = (M / 32) * (N / 32);
  fp8lin_gemm_kernel<<<(tiles + 7) / 8, 256, 0, stream>>>(
      x8, w8, bias, hdr, out, M, N, K);
}
